// Attention_19232863551960
// MI455X (gfx1250) — matrix operand order repaired and checked by reading, not compiled, not run on hardware
//
#include <hip/hip_runtime.h>
#include <hip/hip_bf16.h>

// ---------------------------------------------------------------------------
// Types for CDNA5 WMMA (wave32): v16bf A/B fragments, v8f C/D accumulator
// ---------------------------------------------------------------------------
typedef __attribute__((ext_vector_type(16))) __bf16 v16bf;
typedef __attribute__((ext_vector_type(8)))  __bf16 bf16x8;
typedef __attribute__((ext_vector_type(8)))  unsigned short u16x8;
typedef __attribute__((ext_vector_type(8)))  float  v8f;

static __device__ __forceinline__ v8f wmma_bf16(v16bf a, v16bf b, v8f c) {
  // D = A(16x32 bf16) * B(32x16 bf16) + C(16x16 f32)
  return __builtin_amdgcn_wmma_f32_16x16x32_bf16(
      /*neg_a=*/false, a, /*neg_b=*/false, b,
      /*c_mod=*/(short)0, c, /*reuse_a=*/false, /*reuse_b=*/false);
}

// Raw v_exp_f32 (exp2). Softmax args are <= 0; denorm flush is harmless.
static __device__ __forceinline__ float fast_exp2(float x) {
#if __has_builtin(__builtin_amdgcn_exp2f)
  return __builtin_amdgcn_exp2f(x);
#else
  return exp2f(x);
#endif
}

// Load a 16-element bf16 fragment as two 16B chunks (lo at p, hi at p+gap).
static __device__ __forceinline__ v16bf ldfrag(const __bf16* p, int gap) {
  bf16x8 lo = *(const bf16x8*)(p);
  bf16x8 hi = *(const bf16x8*)(p + gap);
  return __builtin_shufflevector(lo, hi, 0, 1, 2, 3, 4, 5, 6, 7,
                                 8, 9, 10, 11, 12, 13, 14, 15);
}

// Problem constants
#define DMODEL 1024
#define NHEADS 16
#define HEADD  64
#define SEQ    2048
#define BATCH  4
#define ROWS   (BATCH * SEQ)       // 8192
#define QKVN   (3 * DMODEL)        // 3072
#define LOG2E  1.44269504088896340736f
// Q is pre-scaled at the QKV epilogue so S = Q'K^T is already in exp2 domain.
#define QSCALE (0.125f * LOG2E)    // HEAD_DIM^-0.5 * log2(e)

// ---------------------------------------------------------------------------
// Kernel 1: LayerNorm over last dim (1024) + cast to bf16. One block per row.
// ---------------------------------------------------------------------------
__global__ __launch_bounds__(256) void ln_cast_kernel(
    const float* __restrict__ x, const float* __restrict__ w,
    const float* __restrict__ b, __bf16* __restrict__ h) {
  const int row = blockIdx.x;
  const int tid = threadIdx.x;
  const int lane = tid & 31, wave = tid >> 5;
  const float* xr = x + (size_t)row * DMODEL;

  float v[4], s = 0.f, s2 = 0.f;
#pragma unroll
  for (int i = 0; i < 4; ++i) {
    v[i] = xr[tid + i * 256];
    s += v[i];
    s2 += v[i] * v[i];
  }
#pragma unroll
  for (int o = 16; o > 0; o >>= 1) {
    s += __shfl_xor(s, o, 32);
    s2 += __shfl_xor(s2, o, 32);
  }
  __shared__ float ps[8], ps2[8];
  if (lane == 0) { ps[wave] = s; ps2[wave] = s2; }
  __syncthreads();
  s = 0.f; s2 = 0.f;
#pragma unroll
  for (int i = 0; i < 8; ++i) { s += ps[i]; s2 += ps2[i]; }
  const float mu = s * (1.f / DMODEL);
  const float var = s2 * (1.f / DMODEL) - mu * mu;
  const float rstd = rsqrtf(var + 1e-5f);
#pragma unroll
  for (int i = 0; i < 4; ++i) {
    const int c = tid + i * 256;
    const float hv = (v[i] - mu) * rstd * w[c] + b[c];
    h[(size_t)row * DMODEL + c] = (__bf16)hv;
  }
}

// ---------------------------------------------------------------------------
// Kernel 2: f32 -> bf16 cast (weights), grid-stride
// ---------------------------------------------------------------------------
__global__ __launch_bounds__(256) void cast_bf16_kernel(
    const float* __restrict__ in, __bf16* __restrict__ out, int n) {
  for (int i = blockIdx.x * 256 + threadIdx.x; i < n; i += gridDim.x * 256)
    out[i] = (__bf16)in[i];
}

// ---------------------------------------------------------------------------
// Kernel 3/5: tiled bf16 WMMA GEMM.  C[M,Nc] = A[M,K] * Bw[K,Nc]
//   block = 256 threads = 8 waves; tile BM=128, BN=128, BK=32
//   wave owns 2x4 grid of 16x16 C tiles -> 8 WMMAs per K-step
//   Double-buffered LDS: one barrier per K-step, loads overlap WMMA.
//   B tile staged transposed; pairs of k-rows packed into b32 DS stores.
// MODE 0: QKV epilogue -> scatter bf16 into q[b,h,n,d] (pre-scaled by QSCALE),
//         k[b,h,n,d], vT[b,h,d,n]
// MODE 1: f32 epilogue -> Cout
// ---------------------------------------------------------------------------
#define BM 128
#define BN 128
#define BK 32

template <int MODE>
__global__ __launch_bounds__(256) void gemm_wmma_bf16(
    const __bf16* __restrict__ A, const __bf16* __restrict__ Bw,
    int M, int K, int Nc,
    __bf16* __restrict__ qo, __bf16* __restrict__ ko,
    __bf16* __restrict__ vTo, float* __restrict__ Cout) {
  __shared__ __bf16 As[2][BM][BK];   // row-major, 2 x 8KB
  __shared__ __bf16 Bs[2][BN][BK];   // transposed (n-major), 2 x 8KB

  const int tid = threadIdx.x;
  const int wave = tid >> 5, lane = tid & 31;
  const int l15 = lane & 15, kh = lane >> 4;
  const int bm = blockIdx.x * BM;
  const int bn = blockIdx.y * BN;
  const int mt0 = (wave & 3) * 2;    // 2 M tiles
  const int nt0 = (wave >> 2) * 4;   // 4 N tiles

  // Per-thread staging assignment
  const int arow = tid >> 1, ak0 = (tid & 1) * 16;    // A: row, 16 k elems
  const int bkp = (tid & 15) * 2;                     // B: even k row (pair)
  const int bn0 = (tid >> 4) * 8;                     // B: 8 n elems
  const __bf16* aptr = A + (size_t)(bm + arow) * K + ak0;
  const __bf16* bptr = Bw + (size_t)bkp * Nc + bn + bn0;

  v8f acc[2][4] = {};

  // Prologue: stage K-step 0 into buffer 0
  bf16x8 ra0 = *(const bf16x8*)(aptr);
  bf16x8 ra1 = *(const bf16x8*)(aptr + 8);
  u16x8 rb0 = *(const u16x8*)(bptr);        // k row bkp
  u16x8 rb1 = *(const u16x8*)(bptr + Nc);   // k row bkp+1
  *(bf16x8*)(&As[0][arow][ak0])     = ra0;
  *(bf16x8*)(&As[0][arow][ak0 + 8]) = ra1;
#pragma unroll
  for (int j = 0; j < 8; ++j)
    *(unsigned int*)(&Bs[0][bn0 + j][bkp]) =
        (unsigned int)rb0[j] | ((unsigned int)rb1[j] << 16);

  int buf = 0;
  for (int kt = 0; kt < K; kt += BK) {
    __syncthreads();
    const bool has_next = (kt + BK) < K;
    if (has_next) {
      // Issue global loads for the next K-step (overlap with WMMA below)
      const __bf16* an = aptr + kt + BK;
      const __bf16* bp = bptr + (size_t)(kt + BK) * Nc;
      ra0 = *(const bf16x8*)(an);
      ra1 = *(const bf16x8*)(an + 8);
      rb0 = *(const u16x8*)(bp);
      rb1 = *(const u16x8*)(bp + Nc);
      if (kt + 2 * BK < K) {  // warm L2 two steps ahead
        __builtin_prefetch(aptr + kt + 2 * BK, 0, 1);
        __builtin_prefetch(bptr + (size_t)(kt + 2 * BK) * Nc, 0, 1);
      }
    }

    // A fragments: lane row = l15, K-half split by lane[4]
    v16bf af[2];
#pragma unroll
    for (int i = 0; i < 2; ++i)
      af[i] = ldfrag(&As[buf][(mt0 + i) * 16 + l15][kh * 16], 8);
    // B fragments: lane col = l15, 16 contiguous k at kg*16
    v16bf bf_[4];
#pragma unroll
    for (int j = 0; j < 4; ++j)
      bf_[j] = ldfrag(&Bs[buf][(nt0 + j) * 16 + l15][kh * 16], 8);

#pragma unroll
    for (int i = 0; i < 2; ++i)
#pragma unroll
      for (int j = 0; j < 4; ++j)
        acc[i][j] = wmma_bf16(af[i], bf_[j], acc[i][j]);

    if (has_next) {
      // Store staged regs into the other buffer (read next iter after barrier)
      *(bf16x8*)(&As[buf ^ 1][arow][ak0])     = ra0;
      *(bf16x8*)(&As[buf ^ 1][arow][ak0 + 8]) = ra1;
#pragma unroll
      for (int j = 0; j < 8; ++j)
        *(unsigned int*)(&Bs[buf ^ 1][bn0 + j][bkp]) =
            (unsigned int)rb0[j] | ((unsigned int)rb1[j] << 16);
    }
    buf ^= 1;
  }

  // Epilogue. C layout: VGPR r -> row (lane>>4)*8+r, col lane&15
#pragma unroll
  for (int i = 0; i < 2; ++i) {
#pragma unroll
    for (int j = 0; j < 4; ++j) {
      const int gc = bn + (nt0 + j) * 16 + l15;
#pragma unroll
      for (int r = 0; r < 8; ++r) {
        const int gr = bm + (mt0 + i) * 16 + kh * 8 + r;
        const float val = acc[i][j][r];
        if (MODE == 1) {
          Cout[(size_t)gr * Nc + gc] = val;
        } else {
          const int which = gc >> 10;           // 0=q 1=k 2=v
          const int rem = gc & 1023;
          const int head = rem >> 6, d = rem & 63;
          const int bidx = gr >> 11, n = gr & (SEQ - 1);
          const size_t bh = (size_t)(bidx * NHEADS + head);
          if (which == 0)      qo[(bh * SEQ + n) * HEADD + d] = (__bf16)(val * QSCALE);
          else if (which == 1) ko[(bh * SEQ + n) * HEADD + d] = (__bf16)val;
          else                 vTo[(bh * HEADD + d) * SEQ + n] = (__bf16)val;
        }
      }
    }
  }
}

// ---------------------------------------------------------------------------
// Kernel 4: flash attention. Grid = B*H*(SEQ/128); 8 waves/block, each wave
// owns a 16-query tile, streams keys in chunks of 64.
// Q pre-scaled so S is already exp2-domain; raw v_exp_f32 for softmax;
// row sums via ones-column WMMA; K/V fragments preloaded into registers.
// ---------------------------------------------------------------------------
__global__ __launch_bounds__(256) void attn_wmma_kernel(
    const __bf16* __restrict__ q, const __bf16* __restrict__ kbuf,
    const __bf16* __restrict__ vT, __bf16* __restrict__ o) {
  __shared__ __bf16 plds[8][16][64];  // wave-private P repack tiles, 16KB

  const int wave = threadIdx.x >> 5, lane = threadIdx.x & 31;
  const int l15 = lane & 15, kh = lane >> 4;
  const int blk = blockIdx.x;
  const int qb = blk & ((SEQ / 128) - 1);     // query block within sequence
  const int bh = blk >> 4;                    // b*NHEADS + head
  const __bf16* qp = q + (size_t)bh * SEQ * HEADD;
  const __bf16* kp = kbuf + (size_t)bh * SEQ * HEADD;
  const __bf16* vp = vT + (size_t)bh * HEADD * SEQ;

  const int qrow = qb * 128 + wave * 16 + l15;
  // Q A-fragments (16x64 split into two K=32 fragments), loaded once
  v16bf qf[2];
#pragma unroll
  for (int f = 0; f < 2; ++f)
    qf[f] = ldfrag(qp + (size_t)qrow * HEADD + f * 32 + kh * 16, 8);

  // All-ones B fragment: D = P x ones gives per-row sums in every lane
  v16bf ones;
#pragma unroll
  for (int i = 0; i < 16; ++i) ones[i] = (__bf16)1.0f;

  float m[8];
  v8f acc[4] = {};
  v8f accS = {};                      // running row sums (ones-column trick)
#pragma unroll
  for (int r = 0; r < 8; ++r) m[r] = -3.0e38f;

  for (int kc = 0; kc < SEQ; kc += 64) {
    // ---- preload K fragments for 64 keys (8 x b128 pairs, clause) ----
    v16bf kf[4][2];
#pragma unroll
    for (int t = 0; t < 4; ++t) {
      const __bf16* kbp = kp + (size_t)(kc + t * 16 + l15) * HEADD + kh * 16;
      kf[t][0] = ldfrag(kbp, 8);
      kf[t][1] = ldfrag(kbp + 32, 8);
    }
    // ---- S = Q' * K^T, dependent WMMA pairs 4 apart (no hazard NOPs) ----
    v8f st[4];
#pragma unroll
    for (int t = 0; t < 4; ++t) { v8f z = {}; st[t] = wmma_bf16(qf[0], kf[t][0], z); }
#pragma unroll
    for (int t = 0; t < 4; ++t) st[t] = wmma_bf16(qf[1], kf[t][1], st[t]);

    // ---- preload V fragments now; latency hides under softmax VALU ----
    v16bf vf[4][2];
#pragma unroll
    for (int d = 0; d < 4; ++d) {
      const __bf16* vbp = vp + (size_t)(d * 16 + l15) * SEQ + kc + kh * 16;
      vf[d][0] = ldfrag(vbp, 8);
      vf[d][1] = ldfrag(vbp + 32, 8);
    }
    if (kc + 64 < SEQ) {  // warm L2 for next chunk's K and V
      __builtin_prefetch(kp + (size_t)(kc + 64 + l15) * HEADD, 0, 1);
      __builtin_prefetch(vp + (size_t)l15 * SEQ + kc + 64, 0, 1);
    }

    // ---- online softmax (exp2 domain; logits already scaled) ----
    float mt[8];
#pragma unroll
    for (int r = 0; r < 8; ++r)
      mt[r] = fmaxf(fmaxf(st[0][r], st[1][r]), fmaxf(st[2][r], st[3][r]));
#pragma unroll
    for (int off = 8; off > 0; off >>= 1)
#pragma unroll
      for (int r = 0; r < 8; ++r)
        mt[r] = fmaxf(mt[r], __shfl_xor(mt[r], off, 16));
    float alpha[8];
#pragma unroll
    for (int r = 0; r < 8; ++r) {
      const float mn = fmaxf(m[r], mt[r]);
      alpha[r] = fast_exp2(m[r] - mn);
      m[r] = mn;
#pragma unroll
      for (int t = 0; t < 4; ++t) st[t][r] = fast_exp2(st[t][r] - mn);
    }
    // rescale running accumulators (O tiles + row-sum tile)
#pragma unroll
    for (int r = 0; r < 8; ++r) {
      accS[r] *= alpha[r];
#pragma unroll
      for (int d = 0; d < 4; ++d) acc[d][r] *= alpha[r];
    }

    // ---- repack P (C layout) -> A fragment layout via wave-private LDS ----
#pragma unroll
    for (int t = 0; t < 4; ++t)
#pragma unroll
      for (int r = 0; r < 8; ++r)
        plds[wave][kh * 8 + r][t * 16 + l15] = (__bf16)st[t][r];
    asm volatile("" ::: "memory");   // LDS ops are in-order within a wave
    v16bf pf[2];
#pragma unroll
    for (int f = 0; f < 2; ++f)
      pf[f] = ldfrag(&plds[wave][l15][f * 32 + kh * 16], 8);

    // ---- accS += P x ones; O += P x V (dependent pairs 5 apart) ----
    accS = wmma_bf16(pf[0], ones, accS);
#pragma unroll
    for (int d = 0; d < 4; ++d) acc[d] = wmma_bf16(pf[0], vf[d][0], acc[d]);
    accS = wmma_bf16(pf[1], ones, accS);
#pragma unroll
    for (int d = 0; d < 4; ++d) acc[d] = wmma_bf16(pf[1], vf[d][1], acc[d]);
  }

  // ---- normalize and store to attn buffer [b, n, head*64 + d] as bf16 ----
  const int bidx = bh >> 4, head = bh & (NHEADS - 1);
  float inv[8];
#pragma unroll
  for (int r = 0; r < 8; ++r) inv[r] = 1.f / accS[r];
#pragma unroll
  for (int d = 0; d < 4; ++d)
#pragma unroll
    for (int r = 0; r < 8; ++r) {
      const int n = qb * 128 + wave * 16 + kh * 8 + r;
      const int col = head * HEADD + d * 16 + l15;
      o[((size_t)(bidx * SEQ + n)) * DMODEL + col] = (__bf16)(acc[d][r] * inv[r]);
    }
}

// ---------------------------------------------------------------------------
// Launch
// ---------------------------------------------------------------------------
extern "C" void kernel_launch(void* const* d_in, const int* in_sizes, int n_in,
                              void* d_out, int out_size, void* d_ws, size_t ws_size,
                              hipStream_t stream) {
  const float* x     = (const float*)d_in[0];
  const float* ln_w  = (const float*)d_in[1];
  const float* ln_b  = (const float*)d_in[2];
  const float* w_qkv = (const float*)d_in[3];
  const float* w_out = (const float*)d_in[4];
  float* out = (float*)d_out;

  // Workspace carve-up (bf16 buffers); h region is reused for attn output.
  char* ws = (char*)d_ws;
  const size_t sz_h    = (size_t)ROWS * DMODEL * 2;        // 16 MB (h, then attn)
  const size_t sz_wqkv = (size_t)DMODEL * QKVN * 2;        // 6 MB
  const size_t sz_wout = (size_t)DMODEL * DMODEL * 2;      // 2 MB
  const size_t sz_qkv  = (size_t)BATCH * NHEADS * SEQ * HEADD * 2;  // 16 MB each
  __bf16* h    = (__bf16*)(ws);
  __bf16* wqkv = (__bf16*)(ws + sz_h);
  __bf16* wout = (__bf16*)(ws + sz_h + sz_wqkv);
  __bf16* qb   = (__bf16*)(ws + sz_h + sz_wqkv + sz_wout);
  __bf16* kb   = (__bf16*)(ws + sz_h + sz_wqkv + sz_wout + sz_qkv);
  __bf16* vTb  = (__bf16*)(ws + sz_h + sz_wqkv + sz_wout + 2 * sz_qkv);
  __bf16* attn = h;  // reuse: h consumed by QKV GEMM before attn is written

  // 1) LayerNorm + cast
  ln_cast_kernel<<<ROWS, 256, 0, stream>>>(x, ln_w, ln_b, h);
  // 2) weight casts
  cast_bf16_kernel<<<1024, 256, 0, stream>>>(w_qkv, wqkv, DMODEL * QKVN);
  cast_bf16_kernel<<<512, 256, 0, stream>>>(w_out, wout, DMODEL * DMODEL);
  // 3) QKV GEMM (8192x1024 @ 1024x3072) with scatter epilogue
  gemm_wmma_bf16<0><<<dim3(ROWS / BM, QKVN / BN), 256, 0, stream>>>(
      h, wqkv, ROWS, DMODEL, QKVN, qb, kb, vTb, nullptr);
  // 4) flash attention
  attn_wmma_kernel<<<BATCH * NHEADS * (SEQ / 128), 256, 0, stream>>>(
      qb, kb, vTb, attn);
  // 5) output projection (8192x1024 @ 1024x1024) -> f32
  gemm_wmma_bf16<1><<<dim3(ROWS / BM, DMODEL / BN), 256, 0, stream>>>(
      attn, wout, ROWS, DMODEL, DMODEL, nullptr, nullptr, nullptr, out);
}
